// FlashMultiHeadAttention_49409303773595
// MI455X (gfx1250) — matrix operand order repaired and checked by reading, not compiled, not run on hardware
//
#include <hip/hip_runtime.h>
#include <hip/hip_bf16.h>

typedef __attribute__((ext_vector_type(16))) _Float16 v16h;
typedef __attribute__((ext_vector_type(8)))  float    v8f;
typedef int v4i __attribute__((vector_size(16)));   // pointee type for async-LDS builtin

constexpr int B  = 2;
constexpr int S  = 2048;
constexpr int D  = 1024;
constexpr int H  = 16;
constexpr int HD = 64;
constexpr int NKV = 32;                       // kv tile width in flash loop
constexpr float SCALE_LOG2E = 0.18033688f;    // (1/sqrt(64)) * log2(e)

union F16x16 { v16h v; float4 f4[2]; };

__device__ __forceinline__ v8f wmma_f16(v16h a, v16h b, v8f c) {
  return __builtin_amdgcn_wmma_f32_16x16x32_f16(false, a, false, b, (short)0, c,
                                                false, false);
}

// ---------------- async global->LDS staging helpers ----------------

#if __has_builtin(__builtin_amdgcn_global_load_async_to_lds_b128)
#define HAVE_ASYNC_LDS 1
#else
#define HAVE_ASYNC_LDS 0
#endif

__device__ __forceinline__ void async_copy16(const _Float16* g, _Float16* l) {
#if HAVE_ASYNC_LDS
  __builtin_amdgcn_global_load_async_to_lds_b128(
      (__attribute__((address_space(1))) v4i*)g,
      (__attribute__((address_space(3))) v4i*)l, 0, 0);
#else
  *(float4*)l = *(const float4*)g;
#endif
}

__device__ __forceinline__ void wait_async_all() {
#if __has_builtin(__builtin_amdgcn_s_wait_asynccnt)
  __builtin_amdgcn_s_wait_asynccnt(0);
#else
  asm volatile("s_wait_asynccnt 0" ::: "memory");
#endif
}

// Stage a 64-row x 128-half weight chunk (16 KB) into LDS; 256 threads x 4 x 16B.
// g0 points at BT[row0, kc], row stride 1024 halves.  LDS tile row stride 128.
__device__ __forceinline__ void stage_b(const _Float16* __restrict__ g0,
                                        _Float16* l0, int tid) {
#pragma unroll
  for (int i = 0; i < 4; ++i) {
    int c   = tid + i * 256;        // 0..1023 chunk id
    int row = c >> 4;               // 64 rows
    int off = (c & 15) * 8;         // 16 x 16B per row
    async_copy16(g0 + (size_t)row * 1024 + off, l0 + row * 128 + off);
  }
}

// ---------------- WMMA fragment loaders ----------------
// A-matrix 16x32 f16 (MxK): lanes 0-15 -> row=lane, K ko..ko+7 / ko+16..ko+23 (ko=0);
// lanes 16-31 same rows with ko=8.
__device__ __forceinline__ v16h load_a_frag(const _Float16* __restrict__ base,
                                            int ld, int k0, int lane) {
  int r  = lane & 15;
  int ko = (lane & 16) ? 8 : 0;
  const _Float16* p = base + (size_t)r * ld + k0 + ko;
  F16x16 a;
  a.f4[0] = *(const float4*)(p);
  a.f4[1] = *(const float4*)(p + 16);
  return a.v;
}

// B-matrix 32x16 f16 (KxN) from BT[n,k] row-major: lanes 0-15 hold K 0..7,16..23 (col=lane),
// lanes 16-31 hold K 8..15,24..31.
__device__ __forceinline__ v16h load_b_frag(const _Float16* __restrict__ bt,
                                            int ld, int n0, int k0, int lane) {
  int c  = lane & 15;
  int ko = (lane & 16) ? 8 : 0;
  const _Float16* p = bt + (size_t)(n0 + c) * ld + k0 + ko;
  F16x16 b;
  b.f4[0] = *(const float4*)(p);
  b.f4[1] = *(const float4*)(p + 16);
  return b.v;
}

// Same B layout but reading the 64x128 LDS tile (row stride 128) -> ds_load_b128.
__device__ __forceinline__ v16h load_b_frag_lds(const _Float16* lds,
                                                int n0, int k, int lane) {
  int c  = lane & 15;
  int ko = (lane & 16) ? 8 : 0;
  const _Float16* p = lds + (n0 + c) * 128 + k + ko;
  F16x16 b;
  b.f4[0] = *(const float4*)(p);
  b.f4[1] = *(const float4*)(p + 16);
  return b.v;
}

// ---------------- conversion kernels ----------------

__global__ void __launch_bounds__(256) cast_f32_f16(const float* __restrict__ src,
                                                    _Float16* __restrict__ dst, int n) {
  int i = (blockIdx.x * blockDim.x + threadIdx.x) * 4;
  if (i + 3 < n) {
    float4 v = *(const float4*)(src + i);
    union { _Float16 h4[4]; float2 f2; } pk;
    pk.h4[0] = (_Float16)v.x; pk.h4[1] = (_Float16)v.y;
    pk.h4[2] = (_Float16)v.z; pk.h4[3] = (_Float16)v.w;
    *(float2*)(dst + i) = pk.f2;
  }
}

// dst[n*1024 + k] = (f16) src[k*1024 + n]   (1024x1024 transpose via LDS tile)
__global__ void __launch_bounds__(256) transpose_cast_1024(const float* __restrict__ src,
                                                           _Float16* __restrict__ dst) {
  __shared__ _Float16 tile[32][33];
  int kb = blockIdx.x * 32;
  int nb = blockIdx.y * 32;
  int tx = threadIdx.x;           // 0..31
  int ty = threadIdx.y;           // 0..7
#pragma unroll
  for (int i = 0; i < 4; ++i)
    tile[ty + i * 8][tx] = (_Float16)src[(size_t)(kb + ty + i * 8) * 1024 + nb + tx];
  __syncthreads();
#pragma unroll
  for (int i = 0; i < 4; ++i)
    dst[(size_t)(nb + ty + i * 8) * 1024 + kb + tx] = tile[tx][ty + i * 8];
}

// ---------------- QKV projection ----------------
// Block = 8 waves sharing (m, b, h); each wave owns a 16(M) x 64(N) tile.
// Weight slab chunks staged to LDS asynchronously, double buffered.
// q,k stored [B,H,S,HD] f16; v stored transposed [B,H,HD,S] f16.
__global__ void __launch_bounds__(256) gemm_qkv(
    const _Float16* __restrict__ x16,
    const _Float16* __restrict__ Wqt, const _Float16* __restrict__ Wkt,
    const _Float16* __restrict__ Wvt,
    const float* __restrict__ bq, const float* __restrict__ bk,
    const float* __restrict__ bv,
    _Float16* __restrict__ qh, _Float16* __restrict__ kh,
    _Float16* __restrict__ vt) {
  __shared__ _Float16 bsl[2][64 * 128];        // 2 x 16 KB double buffer
  int tid   = threadIdx.x;
  int wslot = tid >> 5;
  int lane  = tid & 31;
  int blk = blockIdx.x;
  int stg = blk & 15;             // group of 8 seq tiles
  int h   = (blk >> 4) & 15;
  int b   = (blk >> 8) & 1;
  int m   = blk >> 9;             // 0=q 1=k 2=v
  int st  = stg * 8 + wslot;

  const _Float16* Wt   = (m == 0) ? Wqt : (m == 1) ? Wkt : Wvt;
  const float*    bias = (m == 0) ? bq  : (m == 1) ? bk  : bv;

  const _Float16* A  = x16 + (size_t)(b * S + st * 16) * D;
  const _Float16* Bg = Wt + (size_t)(h * HD) * D;    // 64 x 1024 slab

  v8f acc[4] = {};

  stage_b(Bg, &bsl[0][0], tid);       // preload chunk 0
  wait_async_all();
  __syncthreads();

  int buf = 0;
  for (int kc = 0; kc < D; kc += 128) {
    if (kc + 128 < D) stage_b(Bg + kc + 128, &bsl[buf ^ 1][0], tid);  // prefetch

    v16h af[4];                        // chunk-batched A fragments (one load clause)
#pragma unroll
    for (int kk = 0; kk < 4; ++kk) af[kk] = load_a_frag(A, D, kc + kk * 32, lane);

#pragma unroll
    for (int kk = 0; kk < 4; ++kk) {
      v16h bm[4];                      // batch B fragments: one ds clause, one wait
#pragma unroll
      for (int nt = 0; nt < 4; ++nt)
        bm[nt] = load_b_frag_lds(&bsl[buf][0], nt * 16, kk * 32, lane);
#pragma unroll
      for (int nt = 0; nt < 4; ++nt)   // 4 independent back-to-back WMMAs
        acc[nt] = wmma_f16(af[kk], bm[nt], acc[nt]);
    }
    if (kc + 128 < D) {
      wait_async_all();
      __syncthreads();
      buf ^= 1;
    }
  }

  int col = lane & 15;
  int rb  = (lane & 16) ? 8 : 0;       // C layout: VGPR j -> row j (+8 hi lanes)
  size_t bh = (size_t)(b * H + h);
#pragma unroll
  for (int nt = 0; nt < 4; ++nt) {
    float bv_ = bias[h * HD + nt * 16 + col];
    if (m < 2) {
      _Float16* dst = ((m == 0) ? qh : kh) + (bh * S + st * 16) * HD + nt * 16 + col;
#pragma unroll
      for (int j = 0; j < 8; ++j)
        dst[(size_t)(rb + j) * HD] = (_Float16)(acc[nt][j] + bv_);
    } else {
      union { _Float16 h8[8]; float4 f4; } pk;
#pragma unroll
      for (int j = 0; j < 8; ++j) pk.h8[j] = (_Float16)(acc[nt][j] + bv_);
      _Float16* dst = vt + (bh * HD + nt * 16 + col) * (size_t)S + st * 16 + rb;
      *(float4*)dst = pk.f4;           // 8 consecutive seq positions
    }
  }
}

// ---------------- flash attention ----------------
// One wave per (b, h, 16-query tile).  Online softmax over 32-wide kv tiles.
__global__ void __launch_bounds__(256) attn(const _Float16* __restrict__ qh,
                                            const _Float16* __restrict__ kh,
                                            const _Float16* __restrict__ vt,
                                            const int* __restrict__ seg,
                                            _Float16* __restrict__ oh) {
  __shared__ _Float16 ldsP[8][16 * 32];     // per-wave P staging (C->A transpose)
  int wslot = threadIdx.x >> 5;
  int lane  = threadIdx.x & 31;
  int wave  = (blockIdx.x * blockDim.x + threadIdx.x) >> 5;
  int qt = wave & 127;
  int h  = (wave >> 7) & 15;
  int b  = (wave >> 11) & 1;

  const _Float16* Q = qh + (size_t)(b * H + h) * S * HD;
  const _Float16* K = kh + (size_t)(b * H + h) * S * HD;
  const _Float16* V = vt + (size_t)(b * H + h) * HD * S;
  const int* segb = seg + b * S;

  int col = lane & 15;
  int rb  = (lane & 16) ? 8 : 0;
  int q0  = qt * 16;

  // Q fragments held across whole kv loop (K-dim = HD = 2 x 32)
  v16h aq0 = load_a_frag(Q + (size_t)q0 * HD, HD, 0, lane);
  v16h aq1 = load_a_frag(Q + (size_t)q0 * HD, HD, 32, lane);

  v8f accO[4] = {};
  float mrow[8], lrow[8];
#pragma unroll
  for (int j = 0; j < 8; ++j) { mrow[j] = -1e30f; lrow[j] = 0.f; }

  int ntile = (q0 + 15) / NKV + 1;   // causal upper bound
  _Float16* lp = ldsP[wslot];

  for (int t = 0; t < ntile; ++t) {
    int kv0 = t * NKV;

    // batch the 4 K fragments (one load clause), then the 4 score WMMAs
    v16h bkf[4];
    bkf[0] = load_b_frag(K, HD, kv0,      0,  lane);
    bkf[1] = load_b_frag(K, HD, kv0,      32, lane);
    bkf[2] = load_b_frag(K, HD, kv0 + 16, 0,  lane);
    bkf[3] = load_b_frag(K, HD, kv0 + 16, 32, lane);
    v8f s0 = {}, s1 = {};
    s0 = wmma_f16(aq0, bkf[0], s0);
    s0 = wmma_f16(aq1, bkf[1], s0);
    s1 = wmma_f16(aq0, bkf[2], s1);
    s1 = wmma_f16(aq1, bkf[3], s1);

    bool kvv0 = segb[kv0 + col]      != -1;
    bool kvv1 = segb[kv0 + 16 + col] != -1;

    float p0[8], p1[8], alpha[8];
#pragma unroll
    for (int j = 0; j < 8; ++j) {
      int qpos = q0 + rb + j;
      float a0 = s0[j] * SCALE_LOG2E;
      float a1 = s1[j] * SCALE_LOG2E;
      a0 = (kvv0 && (kv0 + col)      <= qpos) ? a0 : -1e30f;  // causal+segment
      a1 = (kvv1 && (kv0 + 16 + col) <= qpos) ? a1 : -1e30f;
      float mx = fmaxf(a0, a1);
      mx = fmaxf(mx, __shfl_xor(mx, 1));
      mx = fmaxf(mx, __shfl_xor(mx, 2));
      mx = fmaxf(mx, __shfl_xor(mx, 4));
      mx = fmaxf(mx, __shfl_xor(mx, 8));     // row max over 32 cols
      float mn = fmaxf(mrow[j], mx);
      alpha[j] = exp2f(mrow[j] - mn);
      float e0 = exp2f(a0 - mn);
      float e1 = exp2f(a1 - mn);
      float rs = e0 + e1;
      rs += __shfl_xor(rs, 1);
      rs += __shfl_xor(rs, 2);
      rs += __shfl_xor(rs, 4);
      rs += __shfl_xor(rs, 8);
      lrow[j] = lrow[j] * alpha[j] + rs;
      mrow[j] = mn;
      p0[j] = e0; p1[j] = e1;
    }

#pragma unroll
    for (int nt = 0; nt < 4; ++nt)
#pragma unroll
      for (int j = 0; j < 8; ++j) accO[nt][j] *= alpha[j];

    // P: C-layout -> A-layout via per-wave LDS tile (same-wave LDS is in-order)
#pragma unroll
    for (int j = 0; j < 8; ++j) {
      lp[(rb + j) * 32 + col]      = (_Float16)p0[j];
      lp[(rb + j) * 32 + col + 16] = (_Float16)p1[j];
    }

    // issue V fragment loads now so they fly during the DS round-trip
    v16h bvf[4];
#pragma unroll
    for (int nt = 0; nt < 4; ++nt)
      bvf[nt] = load_b_frag(V, S, nt * 16, kv0, lane);

    asm volatile("s_wait_dscnt 0" ::: "memory");
    int arow = lane & 15;
    int ako  = (lane & 16) ? 8 : 0;
    F16x16 ap;
    ap.f4[0] = *(const float4*)(lp + arow * 32 + ako);
    ap.f4[1] = *(const float4*)(lp + arow * 32 + ako + 16);

#pragma unroll
    for (int nt = 0; nt < 4; ++nt)
      accO[nt] = wmma_f16(ap.v, bvf[nt], accO[nt]);
  }

  // normalize, zero padded queries, store to oh[B*S, H*HD] f16
  float inv_l[8];
#pragma unroll
  for (int j = 0; j < 8; ++j) {
    bool qv = segb[q0 + rb + j] != -1;
    inv_l[j] = (qv && lrow[j] > 0.f) ? (1.0f / lrow[j]) : 0.0f;
  }
#pragma unroll
  for (int nt = 0; nt < 4; ++nt) {
    _Float16* dst = oh + ((size_t)(b * S) + q0) * (H * HD) + h * HD + nt * 16 + col;
#pragma unroll
    for (int j = 0; j < 8; ++j)
      dst[(size_t)(rb + j) * (H * HD)] = (_Float16)(accO[nt][j] * inv_l[j]);
  }
}

// ---------------- output projection ----------------
// Block = 8 waves sharing the 64-col weight slab (async LDS staged, double buffered).
__global__ void __launch_bounds__(256) gemm_out(const _Float16* __restrict__ oh,
                                                const _Float16* __restrict__ Wot,
                                                const float* __restrict__ bo,
                                                float* __restrict__ out) {
  __shared__ _Float16 bsl[2][64 * 128];
  int tid   = threadIdx.x;
  int wslot = tid >> 5;
  int lane  = tid & 31;
  int blk = blockIdx.x;
  int ng  = blk & 15;             // 64-col group
  int mtg = blk >> 4;             // group of 8 row tiles
  int mt  = mtg * 8 + wslot;

  const _Float16* A  = oh + (size_t)(mt * 16) * (H * HD);
  const _Float16* Bg = Wot + (size_t)(ng * 64) * (H * HD);

  v8f acc[4] = {};

  stage_b(Bg, &bsl[0][0], tid);
  wait_async_all();
  __syncthreads();

  int buf = 0;
  for (int kc = 0; kc < H * HD; kc += 128) {
    if (kc + 128 < H * HD) stage_b(Bg + kc + 128, &bsl[buf ^ 1][0], tid);

    v16h af[4];
#pragma unroll
    for (int kk = 0; kk < 4; ++kk) af[kk] = load_a_frag(A, H * HD, kc + kk * 32, lane);

#pragma unroll
    for (int kk = 0; kk < 4; ++kk) {
      v16h bm[4];
#pragma unroll
      for (int nt = 0; nt < 4; ++nt)
        bm[nt] = load_b_frag_lds(&bsl[buf][0], nt * 16, kk * 32, lane);
#pragma unroll
      for (int nt = 0; nt < 4; ++nt)
        acc[nt] = wmma_f16(af[kk], bm[nt], acc[nt]);
    }
    if (kc + 128 < H * HD) {
      wait_async_all();
      __syncthreads();
      buf ^= 1;
    }
  }

  int col = lane & 15;
  int rb  = (lane & 16) ? 8 : 0;
#pragma unroll
  for (int nt = 0; nt < 4; ++nt) {
    int n = ng * 64 + nt * 16 + col;
    float bias_v = bo[n];
    float* dst = out + (size_t)(mt * 16) * D + n;
#pragma unroll
    for (int j = 0; j < 8; ++j)
      dst[(size_t)(rb + j) * D] = acc[nt][j] + bias_v;
  }
}

// ---------------- host launch ----------------
extern "C" void kernel_launch(void* const* d_in, const int* in_sizes, int n_in,
                              void* d_out, int out_size, void* d_ws, size_t ws_size,
                              hipStream_t stream) {
  const float* x  = (const float*)d_in[0];
  const int*   sg = (const int*)d_in[1];
  const float* Wq = (const float*)d_in[2];
  const float* bq = (const float*)d_in[3];
  const float* Wk = (const float*)d_in[4];
  const float* bk = (const float*)d_in[5];
  const float* Wv = (const float*)d_in[6];
  const float* bv = (const float*)d_in[7];
  const float* Wo = (const float*)d_in[8];
  const float* bo = (const float*)d_in[9];
  float* out = (float*)d_out;

  char* ws = (char*)d_ws;                       // 48 MB used, all overwritten per call
  _Float16* x16 = (_Float16*)(ws);              // [B*S, D]      8 MB
  _Float16* Wqt = (_Float16*)(ws + (8  << 20)); // [H*HD, D]     2 MB (transposed)
  _Float16* Wkt = (_Float16*)(ws + (10 << 20));
  _Float16* Wvt = (_Float16*)(ws + (12 << 20));
  _Float16* Wot = (_Float16*)(ws + (14 << 20)); // [D, H*HD]^T
  _Float16* qh  = (_Float16*)(ws + (16 << 20)); // [B,H,S,HD]    8 MB
  _Float16* kh  = (_Float16*)(ws + (24 << 20)); // [B,H,S,HD]
  _Float16* vt  = (_Float16*)(ws + (32 << 20)); // [B,H,HD,S]
  _Float16* oh  = (_Float16*)(ws + (40 << 20)); // [B*S, H*HD]

  cast_f32_f16<<<(B * S * D) / (256 * 4), 256, 0, stream>>>(x, x16, B * S * D);

  dim3 tb(32, 8), tg(32, 32);
  transpose_cast_1024<<<tg, tb, 0, stream>>>(Wq, Wqt);
  transpose_cast_1024<<<tg, tb, 0, stream>>>(Wk, Wkt);
  transpose_cast_1024<<<tg, tb, 0, stream>>>(Wv, Wvt);
  transpose_cast_1024<<<tg, tb, 0, stream>>>(Wo, Wot);

  // 3 * B * H * (S/16/8) = 1536 blocks of 8 waves
  gemm_qkv<<<1536, 256, 0, stream>>>(x16, Wqt, Wkt, Wvt, bq, bk, bv, qh, kh, vt);
  // B * H * (S/16) = 4096 waves
  attn<<<512, 256, 0, stream>>>(qh, kh, vt, sg, oh);
  // (B*S/16/8) * (D/64) = 512 blocks
  gemm_out<<<512, 256, 0, stream>>>(oh, Wot, bo, out);
}
